// BidaF_9801115370128
// MI455X (gfx1250) — matrix operand order repaired and checked by reading, not compiled, not run on hardware
//
#include <hip/hip_runtime.h>
#include <hip/hip_bf16.h>
#include <stdint.h>

// ---------------------------------------------------------------------------
// Types for CDNA5 WMMA (wave32): v_wmma_f32_16x16x32_bf16
// ---------------------------------------------------------------------------
typedef __bf16 v16bf __attribute__((ext_vector_type(16)));
typedef float  v8f   __attribute__((ext_vector_type(8)));

union Frag16 { v16bf v; uint4 q[2]; };

__device__ __forceinline__ unsigned short f2bf_bits(float f) {
  union { float f; unsigned u; } x; x.f = f;
  unsigned r = x.u + 0x7FFFu + ((x.u >> 16) & 1u);   // round-to-nearest-even
  return (unsigned short)(r >> 16);
}
__device__ __forceinline__ void store4_bf16(unsigned short* dst, float a, float b, float c, float d) {
  uint2 u;
  u.x = (unsigned)f2bf_bits(a) | ((unsigned)f2bf_bits(b) << 16);
  u.y = (unsigned)f2bf_bits(c) | ((unsigned)f2bf_bits(d) << 16);
  *(uint2*)dst = u;
}
__device__ __forceinline__ float sigm(float x) { return 1.0f / (1.0f + expf(-x)); }

// A-fragment (16xK tile, M rows): lane m<16 holds K{0..7,16..23}, lane m+16 holds K{8..15,24..31}
__device__ __forceinline__ v16bf load_a_frag(const unsigned short* base, int rs, int lane) {
  const uint4* p = (const uint4*)(base + (size_t)(lane & 15) * rs);
  Frag16 f;
  if (lane < 16) { f.q[0] = p[0]; f.q[1] = p[2]; }
  else           { f.q[0] = p[1]; f.q[1] = p[3]; }
  return f.v;
}
// B-fragment from W stored [N][K] (so B[k][n]=W[n][k]): lane n<16 holds K{0..7,16..23}, lane n+16 K{8..15,24..31}
__device__ __forceinline__ v16bf load_b_frag(const unsigned short* base, int rs, int lane) {
  const uint4* p = (const uint4*)(base + (size_t)(lane & 15) * rs);
  Frag16 f;
  if (lane < 16) { f.q[0] = p[0]; f.q[1] = p[2]; }
  else           { f.q[0] = p[1]; f.q[1] = p[3]; }
  return f.v;
}

// ---------------------------------------------------------------------------
// WMMA GEMM: Y[M,N] = act( X[M,K] @ W[N,K]^T + b1 + b2 )
// block = 256 threads (8 waves); block tile M=64 x N=128; wave grid 4(M) x 2(N);
// each wave computes a 16x64 strip = 4 WMMAs per 32-wide K chunk.
// M must be a multiple of 64; N,K arbitrary (guarded / zero-padded).
// ---------------------------------------------------------------------------
__global__ __launch_bounds__(256) void gemm_bf16(
    const float* __restrict__ X, const float* __restrict__ W,
    const float* __restrict__ bias1, const float* __restrict__ bias2,
    float* __restrict__ Y, int M, int N, int K, int do_relu)
{
  __shared__ __align__(16) unsigned short As[64 * 32];
  __shared__ __align__(16) unsigned short Bs[128 * 32];
  const int tid  = threadIdx.x;
  const int wave = tid >> 5, lane = tid & 31;
  const int wm   = wave >> 1;        // 0..3  M sub-tile
  const int wn   = wave & 1;         // 0..1  N 64-half
  const int m0   = blockIdx.y << 6;
  const int n0b  = blockIdx.x << 7;

  v8f acc[4];
  #pragma unroll
  for (int i = 0; i < 4; ++i) acc[i] = v8f{0.f, 0.f, 0.f, 0.f, 0.f, 0.f, 0.f, 0.f};

  for (int k0 = 0; k0 < K; k0 += 32) {
    const bool fullk = (k0 + 32 <= K);
    // A tile: 64 rows x 32, float4-vectorized (512 float4 slots / 256 threads)
    for (int s = tid; s < 64 * 8; s += 256) {
      const int row = s >> 3, k = (s & 7) << 2;
      const float* xr = X + (size_t)(m0 + row) * K + k0 + k;
      float a0, a1, a2, a3;
      if (fullk) {
        float4 v = *(const float4*)xr;
        a0 = v.x; a1 = v.y; a2 = v.z; a3 = v.w;
        __builtin_prefetch(xr + 32, 0, 0);   // global_prefetch_b8: next K chunk
      } else {
        const int kb = k0 + k;
        a0 = (kb + 0 < K) ? xr[0] : 0.f;
        a1 = (kb + 1 < K) ? xr[1] : 0.f;
        a2 = (kb + 2 < K) ? xr[2] : 0.f;
        a3 = (kb + 3 < K) ? xr[3] : 0.f;
      }
      store4_bf16(As + row * 32 + k, a0, a1, a2, a3);
    }
    // B tile: 128 rows x 32 (1024 float4 slots / 256 threads)
    for (int s = tid; s < 128 * 8; s += 256) {
      const int row = s >> 3, k = (s & 7) << 2;
      const int gn = n0b + row;
      float a0 = 0.f, a1 = 0.f, a2 = 0.f, a3 = 0.f;
      if (gn < N) {
        const float* wr = W + (size_t)gn * K + k0 + k;
        if (fullk) {
          float4 v = *(const float4*)wr;
          a0 = v.x; a1 = v.y; a2 = v.z; a3 = v.w;
          __builtin_prefetch(wr + 32, 0, 0);
        } else {
          const int kb = k0 + k;
          a0 = (kb + 0 < K) ? wr[0] : 0.f;
          a1 = (kb + 1 < K) ? wr[1] : 0.f;
          a2 = (kb + 2 < K) ? wr[2] : 0.f;
          a3 = (kb + 3 < K) ? wr[3] : 0.f;
        }
      }
      store4_bf16(Bs + row * 32 + k, a0, a1, a2, a3);
    }
    __syncthreads();
    v16bf a = load_a_frag(As + wm * (16 * 32), 32, lane);
    #pragma unroll
    for (int nt = 0; nt < 4; ++nt) {
      v16bf b = load_b_frag(Bs + (wn * 64 + nt * 16) * 32, 32, lane);
      acc[nt] = __builtin_amdgcn_wmma_f32_16x16x32_bf16(false, a, false, b, (short)0, acc[nt], false, false);
    }
    __syncthreads();
  }

  const int mbase = m0 + (wm << 4) + ((lane >> 4) << 3);
  #pragma unroll
  for (int nt = 0; nt < 4; ++nt) {
    const int n = n0b + (wn << 6) + (nt << 4) + (lane & 15);
    if (n < N) {
      const float bsum = (bias1 ? bias1[n] : 0.f) + (bias2 ? bias2[n] : 0.f);
      #pragma unroll
      for (int r = 0; r < 8; ++r) {
        float v = acc[nt][r] + bsum;
        if (do_relu) v = fmaxf(v, 0.f);
        Y[(size_t)(mbase + r) * N + n] = v;
      }
    }
  }
}

// ---------------------------------------------------------------------------
// Pack wh [800,200] f32 -> [800,224] bf16 (K zero-padded to 7*32)
// ---------------------------------------------------------------------------
#define KP 224
__global__ void pack_wh_bf16(const float* __restrict__ wh, unsigned short* __restrict__ out) {
  int i = blockIdx.x * blockDim.x + threadIdx.x;
  if (i >= 800 * KP) return;
  int n = i / KP, k = i - n * KP;
  out[i] = (k < 200) ? f2bf_bits(wh[n * 200 + k]) : (unsigned short)0;
}

// ---------------------------------------------------------------------------
// LSTM recurrence, one direction, batched over B=32 sequences (one block).
// Per step: gates[32,800] = h[32,200] @ wh^T via WMMA (M=32 -> 2 tiles,
// N=800 -> 50 tiles over 8 waves, K padded 224 -> 7 chunks), then cell update.
// xw already contains x@wi^T + bi + bh. out gets h at [row,col_off..col_off+200).
// ---------------------------------------------------------------------------
__global__ __launch_bounds__(256) void lstm_rec(
    const float* __restrict__ xw,            // [rows, 800]
    const unsigned short* __restrict__ whb,  // [800, 224] bf16
    float* __restrict__ out,                 // [rows, 400]
    float* __restrict__ gbuf,                // [32*800] global scratch
    int row_base, int T, int col_off, int reverse)
{
  __shared__ __align__(16) unsigned short h_bf[32 * KP];
  __shared__ float c_st[32 * 200];
  const int tid = threadIdx.x;
  const int wave = tid >> 5, lane = tid & 31;
  for (int i = tid; i < 32 * KP; i += 256) h_bf[i] = 0;
  for (int i = tid; i < 32 * 200; i += 256) c_st[i] = 0.f;
  __syncthreads();

  for (int s = 0; s < T; ++s) {
    const int t = reverse ? (T - 1 - s) : s;
    v8f acc[2][7];
    #pragma unroll
    for (int mt = 0; mt < 2; ++mt)
      #pragma unroll
      for (int nt = 0; nt < 7; ++nt)
        acc[mt][nt] = v8f{0.f, 0.f, 0.f, 0.f, 0.f, 0.f, 0.f, 0.f};

    for (int kc = 0; kc < 7; ++kc) {
      const int k0 = kc << 5;
      v16bf a0 = load_a_frag(h_bf + k0, KP, lane);            // batches 0..15
      v16bf a1 = load_a_frag(h_bf + 16 * KP + k0, KP, lane);  // batches 16..31
      #pragma unroll
      for (int nt = 0; nt < 7; ++nt) {
        const int n0 = wave * 112 + (nt << 4);
        if (n0 < 800) {
          const unsigned short* wb = whb + (size_t)n0 * KP + k0;
          if (kc < 6) __builtin_prefetch(wb + 32, 0, 0);      // next K chunk of wh
          v16bf b = load_b_frag(wb, KP, lane);
          acc[0][nt] = __builtin_amdgcn_wmma_f32_16x16x32_bf16(false, a0, false, b, (short)0, acc[0][nt], false, false);
          acc[1][nt] = __builtin_amdgcn_wmma_f32_16x16x32_bf16(false, a1, false, b, (short)0, acc[1][nt], false, false);
        }
      }
    }
    { // spill gate tiles to global scratch (cross-wave gather needed for i/f/g/o)
      const int col = lane & 15, rb = (lane >> 4) << 3;
      for (int nt = 0; nt < 7; ++nt) {
        const int n0 = wave * 112 + (nt << 4);
        if (n0 < 800) {
          #pragma unroll
          for (int mt = 0; mt < 2; ++mt)
            #pragma unroll
            for (int r = 0; r < 8; ++r)
              gbuf[(size_t)(mt * 16 + rb + r) * 800 + n0 + col] = acc[mt][nt][r];
        }
      }
    }
    __threadfence_block();
    __syncthreads();

    for (int u = tid; u < 32 * 200; u += 256) {
      const int b = u / 200, d = u - b * 200;
      const size_t row = (size_t)row_base + (size_t)b * T + t;
      const float* xr = xw + row * 800;
      const float* gr = gbuf + (size_t)b * 800;
      float gi = gr[d]       + xr[d];
      float gf = gr[200 + d] + xr[200 + d];
      float gg = gr[400 + d] + xr[400 + d];
      float go = gr[600 + d] + xr[600 + d];
      float c = sigm(gf) * c_st[u] + sigm(gi) * tanhf(gg);
      c_st[u] = c;
      float h = sigm(go) * tanhf(c);
      h_bf[b * KP + d] = f2bf_bits(h);
      out[row * 400 + col_off + d] = h;
    }
    __threadfence_block();
    __syncthreads();
  }
}

// ---------------------------------------------------------------------------
// Word gather + char-CNN embedding: x[row, 0:100] = word_vec[id],
// x[row, 100:200] = max_pos relu(conv(chars) + b)
// ---------------------------------------------------------------------------
__global__ __launch_bounds__(128) void embed_char(
    const int* __restrict__ word_ids, const int* __restrict__ char_ids,
    const float* __restrict__ word_vec, const float* __restrict__ char_table,
    const float* __restrict__ conv_w, const float* __restrict__ conv_b,
    float* __restrict__ xout, int row_offset)
{
  const int row = blockIdx.x;
  const int tid = threadIdx.x;
  __shared__ float e[16 * 16];
  float* xr = xout + (size_t)(row + row_offset) * 200;
  if (tid < 100) {
    int w = word_ids[row];
    xr[tid] = word_vec[(size_t)w * 100 + tid];
  }
  for (int i = tid; i < 256; i += 128) {
    int pos = i >> 4, cd = i & 15;
    int cid = char_ids[(size_t)row * 16 + pos];
    e[i] = char_table[cid * 16 + cd];
  }
  __syncthreads();
  if (tid < 100) {
    const float* wch = conv_w + tid * 80;   // [5][16]
    float best = -1e30f;
    for (int p = 0; p < 12; ++p) {
      float s = conv_b[tid];
      #pragma unroll
      for (int kk = 0; kk < 5; ++kk)
        #pragma unroll
        for (int cd = 0; cd < 16; ++cd)
          s += e[(p + kk) * 16 + cd] * wch[kk * 16 + cd];
      best = fmaxf(best, s);
    }
    xr[100 + tid] = fmaxf(best, 0.f);
  }
}

// ---------------------------------------------------------------------------
// Attention kernels (D2 = 400, Q = 40, C = 256, B = 32)
// ---------------------------------------------------------------------------
__global__ __launch_bounds__(64) void attn_qproj(
    const float* __restrict__ ce_all, const float* __restrict__ att_w,
    float* __restrict__ s2)
{
  int b = blockIdx.x, j = threadIdx.x;
  if (j >= 40) return;
  const float* q = ce_all + (size_t)(8192 + b * 40 + j) * 400;
  float s = 0.f;
  for (int d = 0; d < 400; ++d) s += q[d] * att_w[400 + d];
  s2[b * 40 + j] = s;
}

__global__ __launch_bounds__(128) void attn_S(
    const float* __restrict__ ce_all, const float* __restrict__ att_w,
    const float* __restrict__ s2, float* __restrict__ S, float* __restrict__ z)
{
  const int rc = blockIdx.x;  // b*256 + c
  const int b = rc >> 8;
  const int tid = threadIdx.x;
  __shared__ float cw3[400];
  __shared__ float red[128];
  __shared__ float Sr[40];
  const float* ce = ce_all + (size_t)rc * 400;
  float part = 0.f;
  for (int d = tid; d < 400; d += 128) {
    float cv = ce[d];
    cw3[d] = cv * att_w[800 + d];
    part += cv * att_w[d];
  }
  red[tid] = part; __syncthreads();
  for (int st = 64; st > 0; st >>= 1) { if (tid < st) red[tid] += red[tid + st]; __syncthreads(); }
  float s1 = red[0];
  __syncthreads();
  if (tid < 40) {
    const float* q = ce_all + (size_t)(8192 + b * 40 + tid) * 400;
    float s = 0.f;
    for (int d = 0; d < 400; ++d) s += cw3[d] * q[d];
    s += s1 + s2[b * 40 + tid];
    Sr[tid] = s;
    S[(size_t)rc * 40 + tid] = s;
  }
  __syncthreads();
  if (tid == 0) {
    float m = -1e30f;
    for (int j = 0; j < 40; ++j) m = fmaxf(m, Sr[j]);
    z[rc] = m;
  }
}

__global__ __launch_bounds__(256) void attn_soft(
    const float* __restrict__ S, const float* __restrict__ z,
    const float* __restrict__ ce_all, float* __restrict__ h_att)
{
  const int b = blockIdx.x, c = threadIdx.x;
  __shared__ float red[256];
  __shared__ float bb[256];
  __shared__ float tj[40];
  float zc = z[b * 256 + c];
  red[c] = zc; __syncthreads();
  for (int st = 128; st > 0; st >>= 1) { if (c < st) red[c] = fmaxf(red[c], red[c + st]); __syncthreads(); }
  float mx = red[0]; __syncthreads();
  float e = expf(zc - mx);
  red[c] = e; __syncthreads();
  for (int st = 128; st > 0; st >>= 1) { if (c < st) red[c] += red[c + st]; __syncthreads(); }
  float den = red[0]; __syncthreads();
  bb[c] = e / den;
  __syncthreads();
  if (c < 40) {
    float s = 0.f;
    for (int cc = 0; cc < 256; ++cc) s += bb[cc] * S[((size_t)b * 256 + cc) * 40 + c];
    tj[c] = s;
  }
  __syncthreads();
  for (int d = c; d < 400; d += 256) {
    float s = 0.f;
    #pragma unroll
    for (int j = 0; j < 40; ++j) s += tj[j] * ce_all[(size_t)(8192 + b * 40 + j) * 400 + d];
    h_att[b * 400 + d] = s;
  }
}

__global__ __launch_bounds__(128) void attn_G(
    const float* __restrict__ ce_all, const float* __restrict__ S,
    const float* __restrict__ h_att, float* __restrict__ G)
{
  const int rc = blockIdx.x, b = rc >> 8, tid = threadIdx.x;
  __shared__ float Sr[40];
  if (tid < 40) Sr[tid] = S[(size_t)rc * 40 + tid];
  __syncthreads();
  const float* ce = ce_all + (size_t)rc * 400;
  float* g = G + (size_t)rc * 1600;
  for (int d = tid; d < 400; d += 128) {
    float c2q = 0.f;
    #pragma unroll
    for (int j = 0; j < 40; ++j) c2q += Sr[j] * ce_all[(size_t)(8192 + b * 40 + j) * 400 + d];
    float cv = ce[d];
    g[d] = cv;
    g[400 + d] = c2q;
    g[800 + d] = cv * c2q;
    g[1200 + d] = cv * h_att[b * 400 + d];
  }
}

// ---------------------------------------------------------------------------
// Output head: logits = [G,Mx] @ pw^T + pb, softmax over C per batch
// ---------------------------------------------------------------------------
__global__ __launch_bounds__(256) void out_p(
    const float* __restrict__ G, const float* __restrict__ Mx,
    const float* __restrict__ pw, const float* __restrict__ pb,
    float* __restrict__ outp)
{
  const int b = blockIdx.x, c = threadIdx.x;
  const size_t row = (size_t)b * 256 + c;
  const float* g = G + row * 1600;
  const float* m = Mx + row * 400;
  float s = pb[0];
  for (int k = 0; k < 1600; ++k) s += g[k] * pw[k];
  for (int k = 0; k < 400; ++k)  s += m[k] * pw[1600 + k];
  __shared__ float red[256];
  red[c] = s; __syncthreads();
  for (int st = 128; st > 0; st >>= 1) { if (c < st) red[c] = fmaxf(red[c], red[c + st]); __syncthreads(); }
  float mx = red[0]; __syncthreads();
  float e = expf(s - mx);
  red[c] = e; __syncthreads();
  for (int st = 128; st > 0; st >>= 1) { if (c < st) red[c] += red[c + st]; __syncthreads(); }
  float den = red[0];
  outp[row] = e / den;
}

// ---------------------------------------------------------------------------
// Host launcher
// ---------------------------------------------------------------------------
extern "C" void kernel_launch(void* const* d_in, const int* in_sizes, int n_in,
                              void* d_out, int out_size, void* d_ws, size_t ws_size,
                              hipStream_t stream) {
  (void)in_sizes; (void)n_in; (void)out_size; (void)ws_size;
  // setup_inputs() dict order:
  const int*   ctx_words  = (const int*)d_in[0];
  const int*   q_words    = (const int*)d_in[1];
  const int*   ctx_char   = (const int*)d_in[2];
  const int*   q_char     = (const int*)d_in[3];
  const float* word_vec   = (const float*)d_in[4];
  const float* char_table = (const float*)d_in[5];
  const float* conv_w     = (const float*)d_in[6];
  const float* conv_b     = (const float*)d_in[7];
  const float* hw1_w      = (const float*)d_in[8];
  const float* hw1_b      = (const float*)d_in[9];
  const float* hw2_w      = (const float*)d_in[10];
  const float* hw2_b      = (const float*)d_in[11];
  const float* att_w      = (const float*)d_in[12];
  const float* p1w        = (const float*)d_in[13];
  const float* p1b        = (const float*)d_in[14];
  const float* p2w        = (const float*)d_in[15];
  const float* p2b        = (const float*)d_in[16];
  const float* lp[24];    // emb_{wi,wh,bi,bh}_{f,b}, m1_..., m2_...
  for (int i = 0; i < 24; ++i) lp[i] = (const float*)d_in[17 + i];

  // Workspace layout (floats)
  const size_t N_X   = (size_t)9472 * 200;
  const size_t N_XW  = (size_t)9472 * 800;
  const size_t N_CE  = (size_t)9472 * 400;
  const size_t N_G   = (size_t)8192 * 1600;
  const size_t N_M   = (size_t)8192 * 400;
  float* ws     = (float*)d_ws;
  float* x_all  = ws;                     // e_all after highway
  float* tmp    = x_all + N_X;
  float* xwbuf  = tmp + N_X;
  float* ce_all = xwbuf + N_XW;
  float* Gbuf   = ce_all + N_CE;
  float* m1buf  = Gbuf + N_G;
  float* m2buf  = m1buf + N_M;
  float* Sbuf   = m2buf + N_M;            // 8192*40
  float* zbuf   = Sbuf + (size_t)8192 * 40;
  float* s2buf  = zbuf + 8192;            // 32*40
  float* hatt   = s2buf + 1280;           // 32*400
  float* gscr   = hatt + 12800;           // 32*800
  unsigned short* whb = (unsigned short*)(gscr + 25600); // 800*224 bf16

  // 1) embeddings (ctx rows [0,8192), query rows [8192,9472))
  embed_char<<<32 * 256, 128, 0, stream>>>(ctx_words, ctx_char, word_vec, char_table, conv_w, conv_b, x_all, 0);
  embed_char<<<32 * 40, 128, 0, stream>>>(q_words, q_char, word_vec, char_table, conv_w, conv_b, x_all, 8192);

  // 2) highway (relu both layers); M=9472 -> 148 row-tiles of 64, N=200 -> 2 col-tiles of 128
  gemm_bf16<<<dim3(2, 148), 256, 0, stream>>>(x_all, hw1_w, hw1_b, nullptr, tmp, 9472, 200, 200, 1);
  gemm_bf16<<<dim3(2, 148), 256, 0, stream>>>(tmp, hw2_w, hw2_b, nullptr, x_all, 9472, 200, 200, 1);

  // helper: one BiLSTM layer
  auto bilstm = [&](const float* xin, int rows, int Kin, const float* const* p,
                    float* outbuf, bool has_query) {
    const dim3 ggrid(7, rows / 64);   // N=800 -> 7 col-tiles of 128
    // forward dir
    gemm_bf16<<<ggrid, 256, 0, stream>>>(xin, p[0], p[2], p[3], xwbuf, rows, 800, Kin, 0);
    pack_wh_bf16<<<(800 * KP + 255) / 256, 256, 0, stream>>>(p[1], whb);
    lstm_rec<<<1, 256, 0, stream>>>(xwbuf, whb, outbuf, gscr, 0, 256, 0, 0);
    if (has_query) lstm_rec<<<1, 256, 0, stream>>>(xwbuf, whb, outbuf, gscr, 8192, 40, 0, 0);
    // backward dir
    gemm_bf16<<<ggrid, 256, 0, stream>>>(xin, p[4], p[6], p[7], xwbuf, rows, 800, Kin, 0);
    pack_wh_bf16<<<(800 * KP + 255) / 256, 256, 0, stream>>>(p[5], whb);
    lstm_rec<<<1, 256, 0, stream>>>(xwbuf, whb, outbuf, gscr, 0, 256, 200, 1);
    if (has_query) lstm_rec<<<1, 256, 0, stream>>>(xwbuf, whb, outbuf, gscr, 8192, 40, 200, 1);
  };

  // 3) emb BiLSTM (ctx + query) -> ce_all [9472,400]
  bilstm(x_all, 9472, 200, &lp[0], ce_all, true);

  // 4) attention -> G [8192,1600]
  attn_qproj<<<32, 64, 0, stream>>>(ce_all, att_w, s2buf);
  attn_S<<<8192, 128, 0, stream>>>(ce_all, att_w, s2buf, Sbuf, zbuf);
  attn_soft<<<32, 256, 0, stream>>>(Sbuf, zbuf, ce_all, hatt);
  attn_G<<<8192, 128, 0, stream>>>(ce_all, Sbuf, hatt, Gbuf);

  // 5) m1 BiLSTM (ctx only, K=1600) -> m1buf [8192,400]
  bilstm(Gbuf, 8192, 1600, &lp[8], m1buf, false);

  // 6) m2 BiLSTM (ctx only, K=400) -> m2buf [8192,400]
  bilstm(m1buf, 8192, 400, &lp[16], m2buf, false);

  // 7) output heads -> d_out = [p1 (8192), p2 (8192)]
  float* outp = (float*)d_out;
  out_p<<<32, 256, 0, stream>>>(Gbuf, m1buf, p1w, p1b, outp);
  out_p<<<32, 256, 0, stream>>>(Gbuf, m2buf, p2w, p2b, outp + 8192);
}
